// MessagePassingNet_2413771620852
// MI455X (gfx1250) — matrix operand order repaired and checked by reading, not compiled, not run on hardware
//
#include <hip/hip_runtime.h>
#include <math.h>

#define DIM   64
#define MLP_H 128
#define NN    4096   // DIM*DIM

typedef _Float16 h8   __attribute__((ext_vector_type(8)));
typedef _Float16 h16  __attribute__((ext_vector_type(16)));
typedef float    f8   __attribute__((ext_vector_type(8)));

__device__ __forceinline__ float sigmoidf_(float x) { return 1.0f / (1.0f + __expf(-x)); }

// monotonic float<->uint mapping for atomicMax-based segment max
__device__ __forceinline__ unsigned enc_f(float f) {
    int b = __float_as_int(f);
    return (b < 0) ? ~(unsigned)b : ((unsigned)b | 0x80000000u);
}
__device__ __forceinline__ float dec_f(unsigned u) {
    int b = (u & 0x80000000u) ? (int)(u & 0x7FFFFFFFu) : (int)(~u);
    return __int_as_float(b);
}

// ---------------- generic zero fill ----------------
__global__ void k_zero_f32(float* p, long n) {
    long i = (long)blockIdx.x * blockDim.x + threadIdx.x;
    if (i < n) p[i] = 0.0f;
}
__global__ void k_zero_u32(unsigned* p, long n) {
    long i = (long)blockIdx.x * blockDim.x + threadIdx.x;
    if (i < n) p[i] = 0u;
}

// ---------------- lin0: out = relu(x @ W + b), block = node ----------------
__global__ void k_lin0(const float* __restrict__ x, const float* __restrict__ w,
                       const float* __restrict__ b, float* __restrict__ out,
                       int N, int F) {
    __shared__ float xs[32];
    int n = blockIdx.x, o = threadIdx.x;
    if (o < F) xs[o] = x[(size_t)n * F + o];
    __syncthreads();
    float acc = b[o];
    for (int i = 0; i < F; ++i) acc = fmaf(xs[i], w[i * DIM + o], acc);
    out[(size_t)n * DIM + o] = fmaxf(acc, 0.0f);
}

// ---------------- edge MLP layer1 (fp16 out), block = edge ----------------
__global__ void k_mlp1(const float* __restrict__ ea, const float* __restrict__ w1,
                       const float* __restrict__ b1, _Float16* __restrict__ h1, int E) {
    __shared__ float es[4];
    int e = blockIdx.x, o = threadIdx.x;
    if (o < 4) es[o] = ea[(size_t)e * 4 + o];
    __syncthreads();
    float acc = b1[o];
    for (int i = 0; i < 4; ++i) acc = fmaf(es[i], w1[i * MLP_H + o], acc);
    h1[(size_t)e * MLP_H + o] = (_Float16)fmaxf(acc, 0.0f);
}

// ---------------- transpose+convert W2: [128,4096] f32 -> [4096,128] f16 ----------------
__global__ void k_w2t(const float* __restrict__ w2, _Float16* __restrict__ w2t) {
    int i = blockIdx.x * blockDim.x + threadIdx.x;
    if (i >= NN * MLP_H) return;
    int n = i / MLP_H, k = i % MLP_H;
    w2t[i] = (_Float16)w2[(size_t)k * NN + n];
}

// ---------------- WMMA GEMM: ew[E,4096] = A[E,128]f16 @ W2 + b2, fp32 accum ----------------
// One wave computes a 16(M) x 64(N) strip of C (4 accumulators), K in 4 steps of 32.
// A fragment (contiguous k order, same as B): lane l in group g holds row M=l&15,
//   halves 0..15 = K k0+g*16 .. +15.
// B fragment (same contiguous k order as A): lane group g holds K k0+g*16 .. +15 for N = lane&15,
//   contiguous in the pre-transposed BT[n,k] -> single v16h load.
__global__ void k_ew_gemm(const _Float16* __restrict__ A, const _Float16* __restrict__ BT,
                          const float* __restrict__ bias, _Float16* __restrict__ C, int M) {
    int wave = blockIdx.x * (blockDim.x >> 5) + (threadIdx.x >> 5);
    int lane = threadIdx.x & 31;
    int mtiles = (M + 15) >> 4;
    int mtile = wave >> 6;       // 64 N-groups (of 4 tiles) per M tile
    int nt4   = wave & 63;
    if (mtile >= mtiles) return;
    int row0 = mtile << 4;
    int n0   = nt4 << 6;
    int g = lane >> 4, l = lane & 15;
    int arow = row0 + l; if (arow >= M) arow = M - 1;

    f8 acc0 = {}, acc1 = {}, acc2 = {}, acc3 = {};
    for (int k0 = 0; k0 < MLP_H; k0 += 32) {
        const _Float16* ap = A + (size_t)arow * MLP_H + k0 + g * 16;
        h8 alo = *(const h8*)ap;
        h8 ahi = *(const h8*)(ap + 8);
        h16 av = __builtin_shufflevector(alo, ahi, 0,1,2,3,4,5,6,7,8,9,10,11,12,13,14,15);

        const _Float16* bp = BT + (size_t)(n0 + l) * MLP_H + k0 + g * 16;
        h16 bv0 = *(const h16*)(bp);
        h16 bv1 = *(const h16*)(bp + 16 * MLP_H);
        h16 bv2 = *(const h16*)(bp + 32 * MLP_H);
        h16 bv3 = *(const h16*)(bp + 48 * MLP_H);

        acc0 = __builtin_amdgcn_wmma_f32_16x16x32_f16(false, av, false, bv0, (short)0, acc0, false, false);
        acc1 = __builtin_amdgcn_wmma_f32_16x16x32_f16(false, av, false, bv1, (short)0, acc1, false, false);
        acc2 = __builtin_amdgcn_wmma_f32_16x16x32_f16(false, av, false, bv2, (short)0, acc2, false, false);
        acc3 = __builtin_amdgcn_wmma_f32_16x16x32_f16(false, av, false, bv3, (short)0, acc3, false, false);
    }
    // C/D layout: lane group g, VGPR r -> row M = g*8+r, col N = lane&15
    for (int r = 0; r < 8; ++r) {
        int row = row0 + g * 8 + r;
        if (row >= M) break;
        size_t base = (size_t)row * NN + l;
        C[base + n0 +  0] = (_Float16)(acc0[r] + bias[n0 +  0 + l]);
        C[base + n0 + 16] = (_Float16)(acc1[r] + bias[n0 + 16 + l]);
        C[base + n0 + 32] = (_Float16)(acc2[r] + bias[n0 + 32 + l]);
        C[base + n0 + 48] = (_Float16)(acc3[r] + bias[n0 + 48 + l]);
    }
}

// ---------------- incoming-edge counts ----------------
__global__ void k_cnt(const int* __restrict__ dst, float* __restrict__ cnt, int E) {
    int e = blockIdx.x * blockDim.x + threadIdx.x;
    if (e < E) atomicAdd(&cnt[dst[e]], 1.0f);
}

// ---------------- msg: per-edge matvec + scatter add, block = edge (64 thr) ----------------
// Stage the edge's 64x64 fp16 weight matrix (8 KB) into LDS via CDNA5 async
// global->LDS b128 transfers (ASYNCcnt path, bypasses VGPRs), then do the
// matvec out of LDS. Each thread issues 8 coalesced 16B async chunks.
__global__ void k_msg(const int* __restrict__ src, const int* __restrict__ dst,
                      const float* __restrict__ hbuf, const _Float16* __restrict__ ew,
                      float* __restrict__ agg, int E) {
    __shared__ _Float16 tile[NN];
    __shared__ float a[DIM];
    int e = blockIdx.x, t = threadIdx.x;
    a[t] = hbuf[(size_t)src[e] * DIM + t];

    const char* gbase = (const char*)(ew + (size_t)e * NN);
    unsigned lbase = (unsigned)(uintptr_t)(&tile[0]);
#pragma unroll
    for (int c = 0; c < 8; ++c) {
        unsigned off = (unsigned)(c * 64 + t) * 16u;   // 512 chunks of 16B, coalesced
        unsigned laddr = lbase + off;
        const char* gaddr = gbase + off;
        asm volatile("global_load_async_to_lds_b128 %0, %1, off"
                     :: "v"(laddr), "v"(gaddr) : "memory");
    }
    asm volatile("s_wait_asynccnt 0x0" ::: "memory");
    __syncthreads();

    float m = 0.0f;
#pragma unroll 8
    for (int i = 0; i < DIM; ++i) m = fmaf(a[i], (float)tile[i * DIM + t], m);
    atomicAdd(&agg[(size_t)dst[e] * DIM + t], m);
}

// ---------------- node update: scatter-mean + root + GRU cell, block = node ----------------
__global__ void k_node(float* __restrict__ hbuf, const float* __restrict__ agg,
                       const float* __restrict__ cnt,
                       const float* __restrict__ cr, const float* __restrict__ cb,
                       const float* __restrict__ wi, const float* __restrict__ wh,
                       const float* __restrict__ bi, const float* __restrict__ bh, int N) {
    __shared__ float hs[DIM], ms[DIM];
    int n = blockIdx.x, o = threadIdx.x;
    hs[o] = hbuf[(size_t)n * DIM + o];
    __syncthreads();
    float root = cb[o];
    for (int i = 0; i < DIM; ++i) root = fmaf(hs[i], cr[i * DIM + o], root);
    float c = fmaxf(cnt[n], 1.0f);
    ms[o] = fmaxf(agg[(size_t)n * DIM + o] / c + root, 0.0f);
    __syncthreads();
    float gir = bi[o],        giz = bi[DIM + o],        gin = bi[2 * DIM + o];
    float ghr = bh[o],        ghz = bh[DIM + o],        ghn = bh[2 * DIM + o];
    for (int i = 0; i < DIM; ++i) {
        float mi = ms[i], hi = hs[i];
        gir = fmaf(mi, wi[i * 3 * DIM + o], gir);
        giz = fmaf(mi, wi[i * 3 * DIM + DIM + o], giz);
        gin = fmaf(mi, wi[i * 3 * DIM + 2 * DIM + o], gin);
        ghr = fmaf(hi, wh[i * 3 * DIM + o], ghr);
        ghz = fmaf(hi, wh[i * 3 * DIM + DIM + o], ghz);
        ghn = fmaf(hi, wh[i * 3 * DIM + 2 * DIM + o], ghn);
    }
    float r  = sigmoidf_(gir + ghr);
    float z  = sigmoidf_(giz + ghz);
    float nn = tanhf(gin + r * ghn);
    hbuf[(size_t)n * DIM + o] = (1.0f - z) * nn + z * hs[o];
}

// ---------------- Set2Set LSTM cell, block = graph (64 thr) ----------------
__global__ void k_lstm(const float* __restrict__ q_star, float* __restrict__ hl, float* __restrict__ cl,
                       const float* __restrict__ wi, const float* __restrict__ wh,
                       const float* __restrict__ bi, const float* __restrict__ bh) {
    __shared__ float qs[2 * DIM], hs[DIM];
    int b = blockIdx.x, t = threadIdx.x;
    qs[t] = q_star[b * 2 * DIM + t];
    qs[t + DIM] = q_star[b * 2 * DIM + DIM + t];
    hs[t] = hl[b * DIM + t];
    __syncthreads();
    float gi = bi[t] + bh[t];
    float gf = bi[DIM + t] + bh[DIM + t];
    float gg = bi[2 * DIM + t] + bh[2 * DIM + t];
    float go = bi[3 * DIM + t] + bh[3 * DIM + t];
    for (int i = 0; i < 2 * DIM; ++i) {
        float q = qs[i];
        gi = fmaf(q, wi[i * 4 * DIM + t], gi);
        gf = fmaf(q, wi[i * 4 * DIM + DIM + t], gf);
        gg = fmaf(q, wi[i * 4 * DIM + 2 * DIM + t], gg);
        go = fmaf(q, wi[i * 4 * DIM + 3 * DIM + t], go);
    }
    for (int i = 0; i < DIM; ++i) {
        float h = hs[i];
        gi = fmaf(h, wh[i * 4 * DIM + t], gi);
        gf = fmaf(h, wh[i * 4 * DIM + DIM + t], gf);
        gg = fmaf(h, wh[i * 4 * DIM + 2 * DIM + t], gg);
        go = fmaf(h, wh[i * 4 * DIM + 3 * DIM + t], go);
    }
    float I = sigmoidf_(gi), F = sigmoidf_(gf), G = tanhf(gg), O = sigmoidf_(go);
    float c = F * cl[b * DIM + t] + I * G;
    cl[b * DIM + t] = c;
    hl[b * DIM + t] = O * tanhf(c);
}

// ---------------- attention scores e[n] + segment max (wave per node) ----------------
__global__ void k_attn_e(const float* __restrict__ hbuf, const float* __restrict__ hl,
                         const int* __restrict__ batch, float* __restrict__ en,
                         unsigned* __restrict__ emx, int N) {
    int n = blockIdx.x * (blockDim.x >> 5) + (threadIdx.x >> 5);
    int lane = threadIdx.x & 31;
    if (n >= N) return;
    int b = batch[n];
    float v = hbuf[(size_t)n * DIM + lane]      * hl[b * DIM + lane]
            + hbuf[(size_t)n * DIM + 32 + lane] * hl[b * DIM + 32 + lane];
    for (int off = 16; off; off >>= 1) v += __shfl_down(v, off, 32);
    if (lane == 0) {
        en[n] = v;
        atomicMax(&emx[b], enc_f(v));
    }
}

__global__ void k_emax_dec(const unsigned* __restrict__ emx, float* __restrict__ emax, int B) {
    int b = blockIdx.x * blockDim.x + threadIdx.x;
    if (b < B) { unsigned u = emx[b]; emax[b] = u ? dec_f(u) : 0.0f; }
}

// ---------------- softmax numerator accumulation (wave per node) ----------------
__global__ void k_attn_acc(const float* __restrict__ hbuf, const float* __restrict__ en,
                           const float* __restrict__ emax, const int* __restrict__ batch,
                           float* __restrict__ racc, float* __restrict__ asum, int N) {
    int n = blockIdx.x * (blockDim.x >> 5) + (threadIdx.x >> 5);
    int lane = threadIdx.x & 31;
    if (n >= N) return;
    int b = batch[n];
    float w = __expf(en[n] - emax[b]);
    atomicAdd(&racc[b * DIM + lane],      w * hbuf[(size_t)n * DIM + lane]);
    atomicAdd(&racc[b * DIM + 32 + lane], w * hbuf[(size_t)n * DIM + 32 + lane]);
    if (lane == 0) atomicAdd(&asum[b], w);
}

__global__ void k_qstar(float* __restrict__ q_star, const float* __restrict__ hl,
                        const float* __restrict__ racc, const float* __restrict__ asum) {
    int b = blockIdx.x, t = threadIdx.x;
    if (t < DIM) q_star[b * 2 * DIM + t] = hl[b * DIM + t];
    else         q_star[b * 2 * DIM + t] = racc[b * DIM + (t - DIM)] / fmaxf(asum[b], 1e-16f);
}

// ---------------- output head, block = graph (64 thr) ----------------
__global__ void k_head(const float* __restrict__ q_star, const float* __restrict__ w1,
                       const float* __restrict__ b1, const float* __restrict__ w2,
                       const float* __restrict__ b2, float* __restrict__ y) {
    __shared__ float qs[2 * DIM];
    __shared__ float red[DIM];
    int b = blockIdx.x, t = threadIdx.x;
    qs[t] = q_star[b * 2 * DIM + t];
    qs[t + DIM] = q_star[b * 2 * DIM + DIM + t];
    __syncthreads();
    float acc = b1[t];
    for (int i = 0; i < 2 * DIM; ++i) acc = fmaf(qs[i], w1[i * DIM + t], acc);
    red[t] = fmaxf(acc, 0.0f) * w2[t];
    __syncthreads();
    for (int s = 32; s; s >>= 1) { if (t < s) red[t] += red[t + s]; __syncthreads(); }
    if (t == 0) y[b] = red[0] + b2[0];
}

extern "C" void kernel_launch(void* const* d_in, const int* in_sizes, int n_in,
                              void* d_out, int out_size, void* d_ws, size_t ws_size,
                              hipStream_t stream) {
    const float* x         = (const float*)d_in[0];
    const float* edge_attr = (const float*)d_in[1];
    const int*   edge_index= (const int*)d_in[2];
    const int*   batch     = (const int*)d_in[3];
    const float* lin0_w    = (const float*)d_in[4];
    const float* lin0_b    = (const float*)d_in[5];
    const float* mlp_w1    = (const float*)d_in[6];
    const float* mlp_b1    = (const float*)d_in[7];
    const float* mlp_w2    = (const float*)d_in[8];
    const float* mlp_b2    = (const float*)d_in[9];
    const float* conv_root = (const float*)d_in[10];
    const float* conv_bias = (const float*)d_in[11];
    const float* gru_wi    = (const float*)d_in[12];
    const float* gru_wh    = (const float*)d_in[13];
    const float* gru_bi    = (const float*)d_in[14];
    const float* gru_bh    = (const float*)d_in[15];
    const float* lstm_wi   = (const float*)d_in[16];
    const float* lstm_wh   = (const float*)d_in[17];
    const float* lstm_bi   = (const float*)d_in[18];
    const float* lstm_bh   = (const float*)d_in[19];
    const float* lin1_w    = (const float*)d_in[20];
    const float* lin1_b    = (const float*)d_in[21];
    const float* lin2_w    = (const float*)d_in[22];
    const float* lin2_b    = (const float*)d_in[23];
    (void)n_in; (void)ws_size;

    const int N = in_sizes[3];
    const int E = in_sizes[1] / 4;
    const int F = in_sizes[0] / N;
    const int B = out_size;

    // workspace partition (256B aligned)
    char* p = (char*)d_ws;
    auto take = [&](size_t bytes) -> char* {
        char* r = p; p += (bytes + 255) & ~(size_t)255; return r;
    };
    float*    hbuf   = (float*)take((size_t)N * DIM * 4);
    _Float16* h1     = (_Float16*)take((size_t)E * MLP_H * 2);
    _Float16* w2t    = (_Float16*)take((size_t)NN * MLP_H * 2);
    _Float16* ew     = (_Float16*)take((size_t)E * NN * 2);       // ~819 MB, fp16
    float*    agg    = (float*)take((size_t)N * DIM * 4);
    float*    cnt    = (float*)take((size_t)N * 4);
    float*    en     = (float*)take((size_t)N * 4);
    float*    q_star = (float*)take((size_t)B * 2 * DIM * 4);
    float*    hl     = (float*)take((size_t)B * DIM * 4);
    float*    cl     = (float*)take((size_t)B * DIM * 4);
    unsigned* emx    = (unsigned*)take((size_t)B * 4);
    float*    emax   = (float*)take((size_t)B * 4);
    float*    asum   = (float*)take((size_t)B * 4);
    float*    racc   = (float*)take((size_t)B * DIM * 4);

    const int* srcI = edge_index;
    const int* dstI = edge_index + E;

    // node embed + edge MLP
    k_lin0<<<N, DIM, 0, stream>>>(x, lin0_w, lin0_b, hbuf, N, F);
    k_mlp1<<<E, MLP_H, 0, stream>>>(edge_attr, mlp_w1, mlp_b1, h1, E);
    k_w2t<<<(NN * MLP_H + 255) / 256, 256, 0, stream>>>(mlp_w2, w2t);
    {
        long waves = (long)((E + 15) / 16) * 64;
        int blocks = (int)((waves + 7) / 8);
        k_ew_gemm<<<blocks, 256, 0, stream>>>(h1, w2t, mlp_b2, ew, E);
    }
    k_zero_f32<<<(N + 255) / 256, 256, 0, stream>>>(cnt, N);
    k_cnt<<<(E + 255) / 256, 256, 0, stream>>>(dstI, cnt, E);

    // 3 message-passing + GRU steps (out == h throughout)
    for (int step = 0; step < 3; ++step) {
        k_zero_f32<<<(int)(((long)N * DIM + 255) / 256), 256, 0, stream>>>(agg, (long)N * DIM);
        k_msg<<<E, DIM, 0, stream>>>(srcI, dstI, hbuf, ew, agg, E);
        k_node<<<N, DIM, 0, stream>>>(hbuf, agg, cnt, conv_root, conv_bias,
                                      gru_wi, gru_wh, gru_bi, gru_bh, N);
    }

    // Set2Set pooling
    k_zero_f32<<<(B * 2 * DIM + 255) / 256, 256, 0, stream>>>(q_star, (long)B * 2 * DIM);
    k_zero_f32<<<(B * DIM + 255) / 256, 256, 0, stream>>>(hl, (long)B * DIM);
    k_zero_f32<<<(B * DIM + 255) / 256, 256, 0, stream>>>(cl, (long)B * DIM);
    for (int it = 0; it < 3; ++it) {
        k_zero_u32<<<(B + 255) / 256, 256, 0, stream>>>(emx, B);
        k_zero_f32<<<(B + 255) / 256, 256, 0, stream>>>(asum, B);
        k_zero_f32<<<(B * DIM + 255) / 256, 256, 0, stream>>>(racc, (long)B * DIM);
        k_lstm<<<B, DIM, 0, stream>>>(q_star, hl, cl, lstm_wi, lstm_wh, lstm_bi, lstm_bh);
        k_attn_e<<<(N + 7) / 8, 256, 0, stream>>>(hbuf, hl, batch, en, emx, N);
        k_emax_dec<<<(B + 255) / 256, 256, 0, stream>>>(emx, emax, B);
        k_attn_acc<<<(N + 7) / 8, 256, 0, stream>>>(hbuf, en, emax, batch, racc, asum, N);
        k_qstar<<<B, 2 * DIM, 0, stream>>>(q_star, hl, racc, asum);
    }

    // output head
    k_head<<<B, DIM, 0, stream>>>(q_star, lin1_w, lin1_b, lin2_w, lin2_b, (float*)d_out);
}
